// DOMINANTBase_58256936403151
// MI455X (gfx1250) — matrix operand order repaired and checked by reading, not compiled, not run on hardware
//
#include <hip/hip_runtime.h>

// ---------------------------------------------------------------------------
// DOMINANT (GCN autoencoder) for MI455X / gfx1250, wave32 + WMMA bf16.
// GEMMs (incl. 10000x10000x128 hs@hs.T) use v_wmma_f32_16x16x32_bf16 with
// double-buffered LDS tiles staged via async global->LDS copies (ASYNCcnt).
// The hs@hs.T result (400MB, stream-once) is stored non-temporally.
// ---------------------------------------------------------------------------

typedef __attribute__((ext_vector_type(16))) __bf16           v16bf;
typedef __attribute__((ext_vector_type(8)))  float            v8f;
typedef __attribute__((ext_vector_type(8)))  unsigned short   u16x8;
typedef __attribute__((ext_vector_type(4)))  int              v4i;

#define IN_DIM 256
#define HID    128
#define TM     128
#define TN     128
#define TK     32

#if __has_builtin(__builtin_amdgcn_global_load_async_to_lds_b128)
#define HAVE_ASYNC_LDS 1
#else
#define HAVE_ASYNC_LDS 0
#endif

union FragU { u16x8 h[2]; v16bf v; };

__device__ __forceinline__ unsigned short f2bf(float f) {
  // round-to-nearest-even f32 -> bf16
  unsigned int u = __float_as_uint(f);
  u += 0x7FFFu + ((u >> 16) & 1u);
  return (unsigned short)(u >> 16);
}

#if HAVE_ASYNC_LDS
typedef __attribute__((address_space(1))) v4i* gv4i_p;   // global int4*
typedef __attribute__((address_space(3))) v4i* lv4i_p;   // LDS int4*

__device__ __forceinline__ void async_cp16(const void* g, void* l) {
  __builtin_amdgcn_global_load_async_to_lds_b128((gv4i_p)g, (lv4i_p)l, 0, 0);
}
__device__ __forceinline__ void wait_async0() {
#if __has_builtin(__builtin_amdgcn_s_wait_asynccnt)
  __builtin_amdgcn_s_wait_asynccnt(0);
#else
  asm volatile("s_wait_asynccnt 0x0" ::: "memory");
#endif
}
#endif

// --------------------------- small prep kernels ----------------------------

__global__ void k_fill1(float* __restrict__ p, int n) {
  int i = blockIdx.x * blockDim.x + threadIdx.x;
  if (i < n) p[i] = 1.0f;                    // self-loop contributes 1 to deg
}

__global__ void k_degacc(const long long* __restrict__ dst, float* __restrict__ deg, int E) {
  int e = blockIdx.x * blockDim.x + threadIdx.x;
  if (e < E) atomicAdd(&deg[(int)dst[e]], 1.0f);
}

__global__ void k_rsqrt(float* __restrict__ p, int n) {
  int i = blockIdx.x * blockDim.x + threadIdx.x;
  if (i < n) p[i] = rsqrtf(p[i]);            // deg >= 1 always (self loop)
}

__global__ void k_cast_bf16(const float* __restrict__ in, unsigned short* __restrict__ out,
                            size_t n, int relu) {
  size_t i = (size_t)blockIdx.x * blockDim.x + threadIdx.x;
  if (i < n) {
    float f = in[i];
    if (relu) f = fmaxf(f, 0.0f);
    out[i] = f2bf(f);
  }
}

// W is [K x Nc] f32 row-major; out is [Nc x K] bf16 row-major (transposed)
__global__ void k_castT_bf16(const float* __restrict__ in, unsigned short* __restrict__ out,
                             int K, int Nc) {
  int i = blockIdx.x * blockDim.x + threadIdx.x;
  if (i < K * Nc) {
    int k = i / Nc, n = i % Nc;
    out[(size_t)n * K + k] = f2bf(in[i]);
  }
}

// agg[i,f] = bias[f] + dis[i]^2 * M[i,f]   (bias + self-loop edge)
__global__ void k_initagg(const float* __restrict__ Mb, const float* __restrict__ dis,
                          const float* __restrict__ bias, float* __restrict__ agg,
                          int n, int F) {
  size_t i = (size_t)blockIdx.x * blockDim.x + threadIdx.x;
  size_t total = (size_t)n * F;
  if (i < total) {
    int r = (int)(i / F);
    int f = (int)(i % F);
    float d = dis[r];
    agg[i] = bias[f] + d * d * Mb[i];
  }
}

// one wave32 per edge: agg[dst,:] += dis[src]*dis[dst] * M[src,:]
__global__ void k_scatter(const float* __restrict__ Mb, const long long* __restrict__ src,
                          const long long* __restrict__ dst, const float* __restrict__ dis,
                          float* __restrict__ agg, int E, int F) {
  int w = blockIdx.x * (blockDim.x >> 5) + (threadIdx.x >> 5);
  int lane = threadIdx.x & 31;
  if (w >= E) return;
  int s = (int)src[w];
  int d = (int)dst[w];
  float nrm = dis[s] * dis[d];
  const float* ms = Mb + (size_t)s * F;
  float* ad = agg + (size_t)d * F;
  for (int f = lane; f < F; f += 32)
    atomicAdd(&ad[f], nrm * ms[f]);
}

// ------------------------------ WMMA GEMM ----------------------------------
// C[M x Nc] f32 = A[M x K] bf16 row-major  X  Bt^T,  Bt is [Nc x K] bf16
// row-major (i.e. C = A @ Bt^T).  256 threads = 8 waves; block tile 128x128;
// K-step 32; each wave computes a 32(M) x 64(N) strip (8 WMMA tiles).
// Double-buffered LDS, async global->LDS staging when available.
// NT != 0: non-temporal C stores (for the stream-once 400MB s_ output).
template <int NT>
__global__ __launch_bounds__(256) void k_gemm_bf16(
    const unsigned short* __restrict__ A, const unsigned short* __restrict__ Bt,
    float* __restrict__ C, int M, int Nc, int K) {
  __shared__ unsigned short lA[2][TM][TK + 8];   // [m][k], 80B row pitch
  __shared__ unsigned short lB[2][TN][TK + 8];   // [n][k] column cache

  const int tid  = threadIdx.x;
  const int lane = tid & 31;
  const int wave = tid >> 5;
  const int half = lane >> 4;
  const int l15  = lane & 15;
  const int wm   = wave & 3;        // M strip: rows wm*32 .. +31
  const int wn   = wave >> 2;       // N strip: cols wn*64 .. +63
  const int m0   = blockIdx.x * TM;
  const int n0   = blockIdx.y * TN;
  const int lrow = tid >> 1;        // 0..127 tile row/col for loading
  const int seg  = tid & 1;         // which 16-element half of the k-slice

  const u16x8 z = {0, 0, 0, 0, 0, 0, 0, 0};

  // stage one k-slice of A and Bt into LDS buffer `buf`
  auto stage = [&](int buf, int kk) {
    int gr = m0 + lrow;
    unsigned short* la = &lA[buf][lrow][seg * 16];
    if (gr < M) {
      const unsigned short* gp = A + (size_t)gr * K + kk + seg * 16;
#if HAVE_ASYNC_LDS
      async_cp16(gp, la);
      async_cp16(gp + 8, la + 8);
#else
      const u16x8* g = (const u16x8*)gp;
      ((u16x8*)la)[0] = g[0];
      ((u16x8*)la)[1] = g[1];
#endif
    } else {
      ((u16x8*)la)[0] = z;
      ((u16x8*)la)[1] = z;
    }
    int gn = n0 + lrow;
    unsigned short* lb = &lB[buf][lrow][seg * 16];
    if (gn < Nc) {
      const unsigned short* gp = Bt + (size_t)gn * K + kk + seg * 16;
#if HAVE_ASYNC_LDS
      async_cp16(gp, lb);
      async_cp16(gp + 8, lb + 8);
#else
      const u16x8* g = (const u16x8*)gp;
      ((u16x8*)lb)[0] = g[0];
      ((u16x8*)lb)[1] = g[1];
#endif
    } else {
      ((u16x8*)lb)[0] = z;
      ((u16x8*)lb)[1] = z;
    }
  };

  v8f acc0[4] = {};
  v8f acc1[4] = {};

  stage(0, 0);
#if HAVE_ASYNC_LDS
  wait_async0();
#endif
  __syncthreads();

  int buf = 0;
  for (int kk = 0; kk < K; kk += TK) {
    if (kk + TK < K) {
      stage(buf ^ 1, kk + TK);          // overlap copy of next slice
    }
#if !HAVE_ASYNC_LDS
    if (kk + TK < K && (m0 + lrow) < M)
      __builtin_prefetch(A + (size_t)(m0 + lrow) * K + kk + TK + seg * 16, 0, 0);
#endif

    // ---- fragments (ISA 7.12.2 layouts; LDS layout makes them contiguous) ----
    FragU fa0, fa1;
    {
      int r0 = wm * 32 + l15;        // subtile 0: lane M = l15
      int r1 = r0 + 16;              // subtile 1
      fa0.h[0] = *(const u16x8*)&lA[buf][r0][half * 8];
      fa0.h[1] = *(const u16x8*)&lA[buf][r0][16 + half * 8];
      fa1.h[0] = *(const u16x8*)&lA[buf][r1][half * 8];
      fa1.h[1] = *(const u16x8*)&lA[buf][r1][16 + half * 8];
    }
#pragma unroll
    for (int ni = 0; ni < 4; ++ni) {
      FragU fb;
      int cn = wn * 64 + ni * 16 + l15;    // lane N = l15
      fb.h[0] = *(const u16x8*)&lB[buf][cn][half * 8];
      fb.h[1] = *(const u16x8*)&lB[buf][cn][16 + half * 8];
      acc0[ni] = __builtin_amdgcn_wmma_f32_16x16x32_bf16(
          false, fa0.v, false, fb.v, (short)0, acc0[ni], false, false);
      acc1[ni] = __builtin_amdgcn_wmma_f32_16x16x32_bf16(
          false, fa1.v, false, fb.v, (short)0, acc1[ni], false, false);
    }

#if HAVE_ASYNC_LDS
    wait_async0();                     // this wave's async writes to LDS done
#endif
    __syncthreads();                   // everyone done reading buf / writing buf^1
    buf ^= 1;
  }

  // ---- store: VGPR r -> M = base + half*8 + r, N = l15 ----
#pragma unroll
  for (int ni = 0; ni < 4; ++ni) {
    int col = n0 + wn * 64 + ni * 16 + l15;
    if (col >= Nc) continue;
    int rb0 = m0 + wm * 32 + half * 8;
    int rb1 = rb0 + 16;
#pragma unroll
    for (int r = 0; r < 8; ++r) {
      int row0 = rb0 + r;
      if (row0 < M) {
        float* p = C + (size_t)row0 * Nc + col;
        if (NT) __builtin_nontemporal_store(acc0[ni][r], p);
        else    *p = acc0[ni][r];
      }
      int row1 = rb1 + r;
      if (row1 < M) {
        float* p = C + (size_t)row1 * Nc + col;
        if (NT) __builtin_nontemporal_store(acc1[ni][r], p);
        else    *p = acc1[ni][r];
      }
    }
  }
}

// ------------------------------- launcher ----------------------------------

extern "C" void kernel_launch(void* const* d_in, const int* in_sizes, int n_in,
                              void* d_out, int out_size, void* d_ws, size_t ws_size,
                              hipStream_t stream) {
  (void)n_in; (void)out_size; (void)ws_size;

  const float*     x    = (const float*)d_in[0];
  const long long* ei   = (const long long*)d_in[1];   // int64 per reference
  const float*     We1  = (const float*)d_in[2];
  const float*     be1  = (const float*)d_in[3];
  const float*     We2  = (const float*)d_in[4];
  const float*     be2  = (const float*)d_in[5];
  const float*     Wa1  = (const float*)d_in[6];
  const float*     ba1  = (const float*)d_in[7];
  const float*     Wa2  = (const float*)d_in[8];
  const float*     ba2  = (const float*)d_in[9];
  const float*     Ws1  = (const float*)d_in[10];
  const float*     bs1  = (const float*)d_in[11];

  const int N = in_sizes[0] / IN_DIM;
  const int E = in_sizes[1] / 2;
  const long long* srcE = ei;
  const long long* dstE = ei + E;

  // ---- deterministic workspace carve (256B aligned) ----
  char* wp = (char*)d_ws;
  auto carve = [&](size_t bytes) -> char* {
    char* p = wp;
    wp += (bytes + 255) & ~(size_t)255;
    return p;
  };
  float*          dis   = (float*)carve((size_t)N * 4);
  unsigned short* Xbf   = (unsigned short*)carve((size_t)N * IN_DIM * 2);
  unsigned short* w1T   = (unsigned short*)carve((size_t)IN_DIM * HID * 2);  // [HID][IN_DIM]
  unsigned short* w2T   = (unsigned short*)carve((size_t)HID * HID * 2);
  unsigned short* wa1T  = (unsigned short*)carve((size_t)HID * HID * 2);
  unsigned short* wa2T  = (unsigned short*)carve((size_t)HID * IN_DIM * 2);  // [IN_DIM][HID]
  unsigned short* ws1T  = (unsigned short*)carve((size_t)HID * HID * 2);
  float*          bufM  = (float*)carve((size_t)N * IN_DIM * 4);   // GEMM result m
  float*          bufA  = (float*)carve((size_t)N * HID * 4);      // aggregated conv
  unsigned short* slotA = (unsigned short*)carve((size_t)N * HID * 2);
  unsigned short* slotB = (unsigned short*)carve((size_t)N * HID * 2);

  float* xout = (float*)d_out;                 // [N, IN_DIM]
  float* sout = xout + (size_t)N * IN_DIM;     // [N, N]

  const int T = 256;
  auto blk1 = [&](size_t n) { return (unsigned)((n + T - 1) / T); };

  auto gemm = [&](const unsigned short* A, const unsigned short* Bt, float* C,
                  int M, int Nc, int K, int nt) {
    dim3 g((M + TM - 1) / TM, (Nc + TN - 1) / TN);
    if (nt) k_gemm_bf16<1><<<g, dim3(T), 0, stream>>>(A, Bt, C, M, Nc, K);
    else    k_gemm_bf16<0><<<g, dim3(T), 0, stream>>>(A, Bt, C, M, Nc, K);
  };
  auto conv = [&](const unsigned short* Hbf, const unsigned short* WT,
                  const float* bias, float* agg, int Fin, int Fout) {
    gemm(Hbf, WT, bufM, N, Fout, Fin, 0);
    k_initagg<<<blk1((size_t)N * Fout), T, 0, stream>>>(bufM, dis, bias, agg, N, Fout);
    k_scatter<<<(E + 7) / 8, T, 0, stream>>>(bufM, srcE, dstE, dis, agg, E, Fout);
  };

  // ---- GCN normalization: dis = rsqrt(deg_with_self_loops) ----
  k_fill1<<<blk1(N), T, 0, stream>>>(dis, N);
  k_degacc<<<blk1(E), T, 0, stream>>>(dstE, dis, E);
  k_rsqrt<<<blk1(N), T, 0, stream>>>(dis, N);

  // ---- cast input to bf16; weights to bf16 transposed ([Nc][K]) ----
  k_cast_bf16<<<blk1((size_t)N * IN_DIM), T, 0, stream>>>(x, Xbf, (size_t)N * IN_DIM, 0);
  k_castT_bf16<<<blk1((size_t)IN_DIM * HID), T, 0, stream>>>(We1, w1T, IN_DIM, HID);
  k_castT_bf16<<<blk1((size_t)HID * HID), T, 0, stream>>>(We2, w2T, HID, HID);
  k_castT_bf16<<<blk1((size_t)HID * HID), T, 0, stream>>>(Wa1, wa1T, HID, HID);
  k_castT_bf16<<<blk1((size_t)HID * IN_DIM), T, 0, stream>>>(Wa2, wa2T, HID, IN_DIM);
  k_castT_bf16<<<blk1((size_t)HID * HID), T, 0, stream>>>(Ws1, ws1T, HID, HID);

  // ---- encoder ----
  conv(Xbf, w1T, be1, bufA, IN_DIM, HID);                                    // conv1
  k_cast_bf16<<<blk1((size_t)N * HID), T, 0, stream>>>(bufA, slotA, (size_t)N * HID, 1); // relu
  conv(slotA, w2T, be2, bufA, HID, HID);                                     // conv2 -> h
  k_cast_bf16<<<blk1((size_t)N * HID), T, 0, stream>>>(bufA, slotB, (size_t)N * HID, 0); // h bf16

  // ---- attribute decoder ----
  conv(slotB, wa1T, ba1, bufA, HID, HID);                                    // conv3
  k_cast_bf16<<<blk1((size_t)N * HID), T, 0, stream>>>(bufA, slotA, (size_t)N * HID, 1); // relu
  conv(slotA, wa2T, ba2, xout, HID, IN_DIM);                                 // conv4 -> x_

  // ---- structure decoder ----
  conv(slotB, ws1T, bs1, bufA, HID, HID);                                    // conv5 -> hs
  k_cast_bf16<<<blk1((size_t)N * HID), T, 0, stream>>>(bufA, slotA, (size_t)N * HID, 0); // hs bf16
  gemm(slotA, slotA, sout, N, N, HID, 1);   // s_ = hs @ hs^T, NT stores (400MB stream)
}
